// SelfA_63333587747382
// MI455X (gfx1250) — matrix operand order repaired and checked by reading, not compiled, not run on hardware
//
#include <hip/hip_runtime.h>

typedef __attribute__((ext_vector_type(16))) _Float16 v16h;
typedef __attribute__((ext_vector_type(8)))  float   v8f;

#define B_    2
#define C_    66
#define T_    32
#define N_    512
#define L_    2
#define QK_   64
#define VCH_  124
#define K_    16
#define CIN_  128
#define NP_   (B_*T_*N_)     /* 32768 points */
#define TN_   (T_*N_)        /* 16384 */

union HF { v16h h; unsigned int u[8]; };

__device__ __forceinline__ void lds_fence() {
  asm volatile("s_wait_dscnt 0" ::: "memory");
}

__device__ __forceinline__ v8f wmma16(const HF& a, const HF& b, v8f c) {
  return __builtin_amdgcn_wmma_f32_16x16x32_f16(false, a.h, false, b.h,
                                                (short)0, c, false, false);
}

// A-fragment (16x32 f16 tile, rows row0..row0+15, cols C0..C0+31) from a
// row-major f16 matrix with row stride strideDw dwords.
// ISA layout: lanes 0-15 -> M=lane, K in {0..7}U{16..23}; lanes 16-31 -> K+8.
__device__ __forceinline__ HF loadA(const unsigned int* p, int strideDw,
                                    int row0, int C0, int m16, int hi) {
  HF a;
  const unsigned int* rp = p + (row0 + m16) * strideDw + (C0 >> 1) + hi * 4;
#pragma unroll
  for (int v = 0; v < 8; ++v) a.u[v] = rp[(v & 3) + ((v >> 2) << 3)];
  return a;
}

// B-fragment (32x16 f16 tile, K rows C0..C0+31) from [n][K] f16 storage with
// row stride strideDw dwords. lanes 0-15: N=lane, K in {0..7}U{16..23}; lanes 16-31: K+8.
__device__ __forceinline__ HF loadB(const unsigned int* p, int strideDw,
                                    int C0, int m16, int hi) {
  HF b;
  const unsigned int* rp = p + m16 * strideDw + (C0 >> 1) + hi * 4;
#pragma unroll
  for (int v = 0; v < 8; ++v) b.u[v] = rp[(v & 3) + ((v >> 2) << 3)];
  return b;
}

__device__ __forceinline__ unsigned long long shfl_xor_u64(unsigned long long v, int m) {
  int lo = (int)(v & 0xffffffffull);
  int hi = (int)(v >> 32);
  lo = __shfl_xor(lo, m, 32);
  hi = __shfl_xor(hi, m, 32);
  return ((unsigned long long)(unsigned int)hi << 32) | (unsigned int)lo;
}

// ---------------- weight conversion: f32 -> f16 (W_v padded to 128 rows) ----
__global__ __launch_bounds__(256) void cvt_w(const float* __restrict__ Wqk,
                                             const float* __restrict__ Wv,
                                             _Float16* __restrict__ wq16,
                                             _Float16* __restrict__ wv16) {
  const int i = blockIdx.x * 256 + threadIdx.x;      // 0 .. 2*128*128-1
  wq16[i] = (_Float16)Wqk[i];
  const int l = i >> 14, rem = i & 16383, o = rem >> 7, c = rem & 127;
  wv16[i] = (o < VCH_) ? (_Float16)Wv[(l * VCH_ + o) * CIN_ + c] : (_Float16)0.f;
}

// ---------------- top-16 neighbor search: one wave per point ----------------
__global__ __launch_bounds__(256) void topk_k(const float* __restrict__ x,
                                              int* __restrict__ idxb) {
  const int lane = threadIdx.x & 31;
  const int p = blockIdx.x * 8 + (threadIdx.x >> 5);  // 0..NP_-1
  const int n = p & (N_ - 1);
  const int t = (p >> 9) & (T_ - 1);
  const int b = p >> 14;
  const float* xb = x + (size_t)b * C_ * TN_;

  const float cx = xb[(0 * T_ + t) * N_ + n];
  const float cy = xb[(1 * T_ + t) * N_ + n];
  const float cz = xb[(2 * T_ + t) * N_ + n];

  float d[48];
#pragma unroll
  for (int s = 0; s < 48; ++s) {
    const int cand = s * 32 + lane;           // 0..1535 (blocks t-1,t,t+1)
    int tj = t - 1 + (cand >> 9);
    tj = tj < 0 ? 0 : (tj > T_ - 1 ? T_ - 1 : tj);
    const int nj = cand & (N_ - 1);
    const float dx = cx - xb[(0 * T_ + tj) * N_ + nj];
    const float dy = cy - xb[(1 * T_ + tj) * N_ + nj];
    const float dz = cz - xb[(2 * T_ + tj) * N_ + nj];
    d[s] = dx * dx + dy * dy + dz * dz;
  }

  unsigned long long selm = 0;   // per-lane "already selected" slot mask
  int keep = 0;
  for (int r = 0; r < K_; ++r) {
    float bd = 3.0e38f;
    int bs = 0;
#pragma unroll
    for (int s = 0; s < 48; ++s) {
      const bool ok = (((selm >> s) & 1ull) == 0) && (d[s] < bd);
      bd = ok ? d[s] : bd;
      bs = ok ? s : bs;
    }
    // lexicographic (dist, candidate-index) key: matches top_k tie-break
    const unsigned int cand = (unsigned int)(bs * 32 + lane);
    unsigned long long key =
        ((unsigned long long)__float_as_uint(bd) << 32) | cand;
#pragma unroll
    for (int m = 1; m < 32; m <<= 1) {
      const unsigned long long o = shfl_xor_u64(key, m);
      key = (o < key) ? o : key;
    }
    const unsigned int wc = (unsigned int)key;       // winning candidate id
    if ((int)(wc & 31u) == lane) selm |= 1ull << (wc >> 5);
    if (r == lane) keep = (int)wc;
  }
  if (lane < K_) idxb[p * K_ + lane] = keep;
}

// ---------------- main fused kernel: one wave per point ---------------------
__global__ __launch_bounds__(256) void attn_main(
    const float* __restrict__ x, const float* __restrict__ bqk,
    const float* __restrict__ bv, const _Float16* __restrict__ wq16,
    const _Float16* __restrict__ wv16, const int* __restrict__ idxb,
    float* __restrict__ out) {
  // per-wave slab: feats [16][128] (2048 halves) + q [16][64] + k [16][64]
  __shared__ _Float16 sm[8 * 4096];   // 64 KB
  const int lane = threadIdx.x & 31;
  const int wv = threadIdx.x >> 5;
  const int hi = lane >> 4;
  const int m16 = lane & 15;

  _Float16* fh = sm + wv * 4096;
  _Float16* qh = fh + 2048;
  _Float16* kh = fh + 3072;
  float* afp = (float*)qh;            // af[16] aliases q slab (dead by then)
  const unsigned int* fdw = (const unsigned int*)fh;
  const unsigned int* qdw = (const unsigned int*)qh;
  const unsigned int* kdw = (const unsigned int*)kh;
  const unsigned int* wqdw_all = (const unsigned int*)wq16;
  const unsigned int* wvdw_all = (const unsigned int*)wv16;

  for (int p = blockIdx.x * 8 + wv; p < NP_; p += 256 * 8) {
    const int n = p & (N_ - 1);
    const int t = (p >> 9) & (T_ - 1);
    const int b = p >> 14;
    const float* xb = x + (size_t)b * C_ * TN_;

    // this lane's neighbor column (j = m16)
    const int cj = idxb[p * K_ + m16];
    int tj = t - 1 + (cj >> 9);
    tj = tj < 0 ? 0 : (tj > T_ - 1 ? T_ - 1 : tj);
    const int nj = cj & (N_ - 1);

    lds_fence();  // previous point's LDS reads are ordered before new writes
    // build feats[j=m16][0..127]: half 'hi' covers source channels [hi*33, hi*33+33)
    {
      const int cbase = hi * 33;
#pragma unroll
      for (int cc = 0; cc < 33; ++cc) {
        const int c = cbase + cc;
        const float xc = xb[(c * T_ + t) * N_ + n];
        const float xn = xb[(c * T_ + tj) * N_ + nj];
        if (c < 4) {
          fh[m16 * 128 + c] = (_Float16)(xc - xn);      // center - neighbor
        } else {
          fh[m16 * 128 + c] = (_Float16)xc;             // center ch 4..65
          fh[m16 * 128 + 62 + c] = (_Float16)xn;        // neighbor -> ch 66..127
        }
      }
    }
    lds_fence();

    for (int l = 0; l < L_; ++l) {
      const unsigned int* wqdw = wqdw_all + l * (128 * 64);
      const unsigned int* wvdw = wvdw_all + l * (128 * 64);

      // ---- qk = W_qk(128x128) * feats(128x16) + b_qk ----
      for (int mt = 0; mt < 8; ++mt) {
        v8f acc = {0.f, 0.f, 0.f, 0.f, 0.f, 0.f, 0.f, 0.f};
#pragma unroll
        for (int ks = 0; ks < 4; ++ks) {
          HF a = loadA(wqdw, 64, mt * 16, ks * 32, m16, hi);
          HF bb = loadB(fdw, 64, ks * 32, m16, hi);
          acc = wmma16(a, bb, acc);
        }
#pragma unroll
        for (int r = 0; r < 8; ++r) {
          const int row = mt * 16 + r + 8 * hi;          // output channel
          const float val = acc[r] + bqk[l * 128 + row];
          if (mt < 4) qh[m16 * 64 + row] = (_Float16)val;        // q[i][c]
          else        kh[m16 * 64 + (row - 64)] = (_Float16)val; // k[j][c]
        }
      }
      lds_fence();

      // ---- energy = q^T k / sqrt(64); softmax over j; af = att[:,0] ----
      v8f e = {0.f, 0.f, 0.f, 0.f, 0.f, 0.f, 0.f, 0.f};
#pragma unroll
      for (int ks = 0; ks < 2; ++ks) {
        HF a = loadA(qdw, 32, 0, ks * 32, m16, hi);   // A[i][c] = q[c,i]
        HF bb = loadB(kdw, 32, ks * 32, m16, hi);     // B[c][j] = k[c,j]
        e = wmma16(a, bb, e);
      }
#pragma unroll
      for (int r = 0; r < 8; ++r) {
        const float ev = e[r] * 0.125f;               // / sqrt(QK)
        float mx = ev;
        mx = fmaxf(mx, __shfl_xor(mx, 1, 32));
        mx = fmaxf(mx, __shfl_xor(mx, 2, 32));
        mx = fmaxf(mx, __shfl_xor(mx, 4, 32));
        mx = fmaxf(mx, __shfl_xor(mx, 8, 32));
        const float pe = __expf(ev - mx);
        float s = pe;
        s += __shfl_xor(s, 1, 32);
        s += __shfl_xor(s, 2, 32);
        s += __shfl_xor(s, 4, 32);
        s += __shfl_xor(s, 8, 32);
        const float p0 = __shfl(pe, lane & 16, 32);   // j==0 holder of this half
        if (m16 == 0) afp[r + 8 * hi] = p0 / s;       // af[i], i = r + 8*hi
      }
      lds_fence();
      const float afL = afp[m16];                     // af for this lane's column

      // ---- v = W_v * feats + b_v ; out[c] = sum_i af[i] * v[c,i] ----
      float* outl = out + (size_t)(l * B_ + b) * 128 * TN_;
      for (int mt = 0; mt < 8; ++mt) {
        v8f acc = {0.f, 0.f, 0.f, 0.f, 0.f, 0.f, 0.f, 0.f};
#pragma unroll
        for (int ks = 0; ks < 4; ++ks) {
          HF a = loadA(wvdw, 64, mt * 16, ks * 32, m16, hi);
          HF bb = loadB(fdw, 64, ks * 32, m16, hi);
          acc = wmma16(a, bb, acc);
        }
#pragma unroll
        for (int r = 0; r < 8; ++r) {
          const int row = mt * 16 + r + 8 * hi;
          const float bvv = (row < VCH_) ? bv[l * VCH_ + row] : 0.f;
          float val = (acc[r] + bvv) * afL;
          val += __shfl_xor(val, 1, 32);
          val += __shfl_xor(val, 2, 32);
          val += __shfl_xor(val, 4, 32);
          val += __shfl_xor(val, 8, 32);
          if (m16 == r && row < VCH_)
            outl[(size_t)(4 + row) * TN_ + t * N_ + n] = val;
        }
      }
      // pos channels 0..3
      if (lane < 4)
        outl[(size_t)lane * TN_ + t * N_ + n] = xb[(lane * T_ + t) * N_ + n];
      lds_fence();  // q/k/af slab rewritten next layer / next point
    }
  }
}

extern "C" void kernel_launch(void* const* d_in, const int* in_sizes, int n_in,
                              void* d_out, int out_size, void* d_ws, size_t ws_size,
                              hipStream_t stream) {
  (void)in_sizes; (void)n_in; (void)out_size; (void)ws_size;
  const float* x   = (const float*)d_in[0];
  const float* Wqk = (const float*)d_in[1];
  const float* bqk = (const float*)d_in[2];
  const float* Wv  = (const float*)d_in[3];
  const float* bv  = (const float*)d_in[4];
  float* out = (float*)d_out;

  char* ws = (char*)d_ws;
  int* idxb = (int*)ws;                                   // 32768*16*4 = 2 MB
  _Float16* wq16 = (_Float16*)(ws + 2097152);             // 64 KB
  _Float16* wv16 = (_Float16*)(ws + 2097152 + 65536);     // 64 KB (padded)

  cvt_w<<<128, 256, 0, stream>>>(Wqk, Wv, wq16, wv16);
  topk_k<<<NP_ / 8, 256, 0, stream>>>(x, idxb);
  attn_main<<<256, 256, 0, stream>>>(x, bqk, bv, wq16, wv16, idxb, out);
}
